// RecurrentNeuralNetwork_773094113978
// MI455X (gfx1250) — matrix operand order repaired and checked by reading, not compiled, not run on hardware
//
#include <hip/hip_runtime.h>
#include <hip/hip_bf16.h>

// RNN forward on MI455X (gfx1250, wave32, WMMA).
// Recurrent GEMM per timestep done with v_wmma_f32_16x16x32_f16,
// W_rec kept as f16 in LDS (320KB/WGP), h state kept fp32 in LDS.

typedef __attribute__((ext_vector_type(16))) _Float16 v16h;
typedef __attribute__((ext_vector_type(8)))  _Float16 v8h;
typedef __attribute__((ext_vector_type(8)))  float    v8f;

#define N_HID   256
#define N_EXC   180
#define BATCH   512
#define TSTEPS  1024
#define BT      16            // batch tile per workgroup
#define LSTR    264           // padded row stride (elements) for LDS tiles

__launch_bounds__(256, 1)
__global__ void rnn_wmma_kernel(const float* __restrict__ in_sig,   // [B,T,3]
                                const float* __restrict__ hid0,     // [B,H]
                                const float* __restrict__ noise,    // [T,B,H]
                                const float* __restrict__ w_in,     // [H,3]
                                const float* __restrict__ w_hh,     // [H,H]
                                const float* __restrict__ w_out,    // [2,H]
                                float* __restrict__ out)            // hidden_list | output_list | hidden_final
{
    __shared__ _Float16 wrec_lds[N_HID * LSTR]; // f16 W_rec (sign folded), row n = output neuron, col k
    __shared__ _Float16 a_lds[BT * LSTR];       // f16 tanh(h) tile  [16 x 256]
    __shared__ float    h_lds[BT * LSTR];       // fp32 hidden state [16 x 256]
    __shared__ float    win_lds[N_HID * 3];
    __shared__ float    wout_lds[2 * N_HID];
    __shared__ float    x_lds[BT * 3];

    const int tid  = threadIdx.x;
    const int b0   = blockIdx.x * BT;           // first batch row of this tile
    const int lane = tid & 31;
    const int wv   = tid >> 5;                  // wave id 0..7
    const int lh   = lane >> 4;                 // lane half (0: lanes 0-15, 1: lanes 16-31)
    const int l16  = lane & 15;
    const int N0c  = wv * 32 + l16;             // this lane's first output column
    const int N1c  = N0c + 16;                  // second output column tile

    const size_t O1 = (size_t)BATCH * TSTEPS * N_HID;          // output_list offset
    const size_t O2 = O1 + (size_t)BATCH * TSTEPS * 2;         // hidden_final offset

    // ---- one-time preload: W_rec (f16, signed columns), W_in, W_out, h0 ----
    for (int i = tid; i < N_HID * N_HID; i += 256) {
        const int k = i & 255;
        float v = w_hh[i];
        if (k >= N_EXC) v = -v;                 // w_rec = w_hh * diag(sign)
        wrec_lds[(i >> 8) * LSTR + k] = (_Float16)v;
    }
    for (int i = tid; i < N_HID * 3; i += 256) win_lds[i]  = w_in[i];
    for (int i = tid; i < 2 * N_HID; i += 256) wout_lds[i] = w_out[i];
    #pragma unroll
    for (int j = 0; j < BT; ++j)
        h_lds[j * LSTR + tid] = hid0[(size_t)(b0 + j) * N_HID + tid];
    __syncthreads();

    // per-lane W_in columns for C-init (invariant over t and r)
    const float wA0 = win_lds[N0c * 3 + 0], wA1 = win_lds[N0c * 3 + 1], wA2 = win_lds[N0c * 3 + 2];
    const float wB0 = win_lds[N1c * 3 + 0], wB1 = win_lds[N1c * 3 + 1], wB2 = win_lds[N1c * 3 + 2];

    for (int t = 0; t < TSTEPS; ++t) {
        // ---- phase 1: x_t load, tanh(h) -> f16 A tile, prefetch next noise tile ----
        if (tid < BT * 3) {
            const int b = tid / 3, k = tid % 3;
            x_lds[tid] = in_sig[((size_t)(b0 + b) * TSTEPS + t) * 3 + k];
        }
        #pragma unroll
        for (int j = 0; j < BT; ++j) {
            const float hv = h_lds[j * LSTR + tid];
            a_lds[j * LSTR + tid] = (_Float16)tanhf(hv);
        }
        {   // prefetch next step's 16KB noise tile (64B per thread)
            const int tn = (t + 1) & (TSTEPS - 1);
            const float* pf = noise + ((size_t)tn * BATCH + b0) * N_HID + (size_t)tid * 16;
            __builtin_prefetch(pf, 0, 0);
        }
        __syncthreads();

        // ---- phase 2: WMMA  tmp = x@W_in^T (C-init)  +  tanh(h) @ W_rec^T ----
        v8f acc0, acc1;
        #pragma unroll
        for (int r = 0; r < 8; ++r) {
            const int M = r + 8 * lh;
            const float x0 = x_lds[M * 3 + 0], x1 = x_lds[M * 3 + 1], x2 = x_lds[M * 3 + 2];
            acc0[r] = x0 * wA0 + x1 * wA1 + x2 * wA2;
            acc1[r] = x0 * wB0 + x1 * wB1 + x2 * wB2;
        }
        #pragma unroll
        for (int ks = 0; ks < 8; ++ks) {
            // A fragment (16x32 f16): lanes 0-15 rows M, K=kA..kA+15
            const int kA = ks * 32 + lh * 16;
            const v8h alo = *(const v8h*)&a_lds[l16 * LSTR + kA];
            const v8h ahi = *(const v8h*)&a_lds[l16 * LSTR + kA + 8];
            const v16h Af = __builtin_shufflevector(alo, ahi, 0,1,2,3,4,5,6,7,8,9,10,11,12,13,14,15);
            // B fragments (32x16 f16): lane covers column N, K=kB..kB+15
            const int kB = ks * 32 + lh * 16;
            const v8h b0lo = *(const v8h*)&wrec_lds[N0c * LSTR + kB];
            const v8h b0hi = *(const v8h*)&wrec_lds[N0c * LSTR + kB + 8];
            const v16h Bf0 = __builtin_shufflevector(b0lo, b0hi, 0,1,2,3,4,5,6,7,8,9,10,11,12,13,14,15);
            acc0 = __builtin_amdgcn_wmma_f32_16x16x32_f16(false, Af, false, Bf0, (short)0, acc0, false, false);
            const v8h b1lo = *(const v8h*)&wrec_lds[N1c * LSTR + kB];
            const v8h b1hi = *(const v8h*)&wrec_lds[N1c * LSTR + kB + 8];
            const v16h Bf1 = __builtin_shufflevector(b1lo, b1hi, 0,1,2,3,4,5,6,7,8,9,10,11,12,13,14,15);
            acc1 = __builtin_amdgcn_wmma_f32_16x16x32_f16(false, Af, false, Bf1, (short)0, acc1, false, false);
        }

        // ---- phase 3: leaky integration + noise, update h, store hidden_list ----
        const float* nzt = noise + ((size_t)t * BATCH + b0) * N_HID;
        #pragma unroll
        for (int r = 0; r < 8; ++r) {
            const int M  = r + 8 * lh;
            const int gb = b0 + M;
            const float nv0 = nzt[M * N_HID + N0c];
            const float nv1 = nzt[M * N_HID + N1c];
            const float h0  = h_lds[M * LSTR + N0c];
            const float h1  = h_lds[M * LSTR + N1c];
            const float hn0 = 0.75f * h0 + 0.25f * acc0[r] + 0.025f * nv0; // 0.025 = sigma*sqrt(alpha)
            const float hn1 = 0.75f * h1 + 0.25f * acc1[r] + 0.025f * nv1;
            h_lds[M * LSTR + N0c] = hn0;
            h_lds[M * LSTR + N1c] = hn1;
            const size_t ob = ((size_t)gb * TSTEPS + t) * N_HID;
            out[ob + N0c] = hn0;
            out[ob + N1c] = hn1;
        }
        __syncthreads();

        // ---- phase 4: output projection out = h @ W_out^T  (256 -> 2) ----
        {
            const int b = tid >> 4, g = tid & 15;
            float s0 = 0.f, s1 = 0.f;
            #pragma unroll
            for (int q = 0; q < 16; ++q) {
                const int n = g * 16 + q;
                const float hv = h_lds[b * LSTR + n];
                s0 += hv * wout_lds[n];
                s1 += hv * wout_lds[N_HID + n];
            }
            #pragma unroll
            for (int off = 8; off > 0; off >>= 1) {
                s0 += __shfl_xor(s0, off, 16);
                s1 += __shfl_xor(s1, off, 16);
            }
            if (g == 0) {
                const size_t oi = O1 + ((size_t)(b0 + b) * TSTEPS + t) * 2;
                out[oi + 0] = s0;
                out[oi + 1] = s1;
            }
        }
        // next iteration's h_lds writes are fenced by its phase-1 barrier
    }

    // ---- hidden_final ----
    #pragma unroll
    for (int j = 0; j < BT; ++j)
        out[O2 + (size_t)(b0 + j) * N_HID + tid] = h_lds[j * LSTR + tid];
}

extern "C" void kernel_launch(void* const* d_in, const int* in_sizes, int n_in,
                              void* d_out, int out_size, void* d_ws, size_t ws_size,
                              hipStream_t stream) {
    const float* in_sig = (const float*)d_in[0];
    const float* hid0   = (const float*)d_in[1];
    const float* noise  = (const float*)d_in[2];
    const float* w_in   = (const float*)d_in[3];
    const float* w_hh   = (const float*)d_in[4];
    const float* w_out  = (const float*)d_in[5];
    float* out = (float*)d_out;
    (void)in_sizes; (void)n_in; (void)out_size; (void)d_ws; (void)ws_size;

    dim3 grid(BATCH / BT);   // 32 workgroups, one 16-row batch tile each
    dim3 block(256);         // 8 wave32 waves
    rnn_wmma_kernel<<<grid, block, 0, stream>>>(in_sig, hid0, noise, w_in, w_hh, w_out, out);
}
